// HyperspectralRegionMoE_50122268344641
// MI455X (gfx1250) — matrix operand order repaired and checked by reading, not compiled, not run on hardware
//
#include <hip/hip_runtime.h>

// ---------------------------------------------------------------------------
// HyperspectralRegionMoE on gfx1250 (MI455X).
// All conv layers run as implicit GEMM on v_wmma_f32_16x16x32_bf16.
// 1x1-conv input tiles staged via global_load_async_to_lds_b128 (ASYNCcnt).
// Activation layout in workspace: channel-major bf16 raw bits: act[c][b*1024+hw]
// ---------------------------------------------------------------------------

#define NPIX 131072                 // 128 * 32 * 32
#define BN_S 0.99999500003749968f   // 1/sqrt(1 + 1e-5)

typedef __attribute__((ext_vector_type(16))) __bf16 v16bf;
typedef __attribute__((ext_vector_type(8)))  float  v8f;

__device__ __forceinline__ unsigned short f2bf(float f) {
  unsigned int u = __builtin_bit_cast(unsigned int, f);
  unsigned int r = u + 0x7FFFu + ((u >> 16) & 1u);   // round-to-nearest-even
  return (unsigned short)(r >> 16);
}
__device__ __forceinline__ float bf2f(unsigned short h) {
  unsigned int u = ((unsigned int)h) << 16;
  return __builtin_bit_cast(float, u);
}

// ---------------------------------------------------------------------------
// Generic WMMA conv kernel.  KS = 1 or 3, CIN = input channels (compile time).
// Block: 128 threads (4 waves).  Block tile: 16 rows x 32 cols of one image.
// grid.x = 256  (128 images * 2 half-images)
// K ordering: KS==1 -> k = ci ; KS==3 -> k = (kh,kw) outer, ci-blocks inner
// (A and B fragments use the same mapping, so the GEMM sum is unchanged).
// Per m-tile, ALL A fragments are hoisted into registers and reused across
// the 8 pixel-subtiles each wave owns; K loop fully unrolled (KSTEPS const).
// Epilogue modes:
//   0: out_bf16 = v + bias
//   1: out_bf16 = leaky(v+bias, 0.01)
//   2: out_bf16 += leaky(v+bias, 0.01)             (residual accumulate)
//   3: t = leaky(bn(v+bias), 0.1) * wgt[b];  out_bf16 = accum ? out+t : t
//   4: out_bf16 = relu(bn(v+bias))                 (projection)
//   5: outF32[NCHW] = relu(bn(v+bias))             (fusion -> d_out)
// ---------------------------------------------------------------------------
template<int KS, int CIN>
__global__ void conv_wmma(const unsigned short* __restrict__ in,
                          const float* __restrict__ W,
                          const float* __restrict__ bias,
                          unsigned short* __restrict__ out,
                          float* __restrict__ outF32,
                          int CinW, int Cout, int mode,
                          const float* __restrict__ bng,
                          const float* __restrict__ bnb,
                          const float* __restrict__ wgt, int accum)
{
  constexpr int PAD    = (KS == 3) ? 1 : 0;
  constexpr int HT     = 16 + 2 * PAD;
  constexpr int WT     = 32 + 2 * PAD;
  constexpr int HTWT   = HT * WT;
  constexpr int KSTEPS = (CIN * KS * KS) / 32;
  constexpr int CIN32  = CIN / 32;
  extern __shared__ unsigned short tile[];

  const int tid   = threadIdx.x;
  const int b     = blockIdx.x >> 1;
  const int r0    = (blockIdx.x & 1) * 16;
  const int nbase = b * 1024;

  // prefetch weights into cache (global_prefetch_b8); 32 floats per cacheline
  for (int o = tid * 32; o < Cout * CinW * KS * KS; o += blockDim.x * 32)
    __builtin_prefetch(W + o, 0, 0);

  // ---- stage input tile into LDS ------------------------------------------
  if constexpr (KS == 1) {
    // No halo: CIN contiguous 1KB row segments -> async DMA straight to LDS.
    // Each iteration moves 8 bf16 (16B) per lane; everything 16B aligned.
    constexpr int CHUNKS = CIN * 64;             // 8-elem chunks in the tile
    for (int c = tid; c < CHUNKS; c += blockDim.x) {
      int ci  = c >> 6;
      int off = (c & 63) << 3;
      unsigned long long g =
          (unsigned long long)(in + (size_t)ci * NPIX + nbase + r0 * 32 + off);
      unsigned int l = (unsigned int)(((ci << 9) + off) * 2);
      asm volatile("global_load_async_to_lds_b128 %0, %1, off"
                   :: "v"(l), "v"(g) : "memory");
    }
    asm volatile("s_wait_asynccnt 0x0" ::: "memory");
  } else {
    // 3x3: halo + zero padding -> scalar staging
    constexpr int ELEMS = CIN * HTWT;
    for (int i = tid; i < ELEMS; i += blockDim.x) {
      int ci  = i / HTWT;
      int rem = i - ci * HTWT;
      int th  = rem / WT;
      int tw  = rem - th * WT;
      int row = r0 + th - PAD;
      int col = tw - PAD;
      unsigned short v = 0;
      if (row >= 0 && row < 32 && col >= 0 && col < 32)
        v = in[ci * NPIX + nbase + row * 32 + col];
      tile[i] = v;
    }
  }
  __syncthreads();

  const int lane = tid & 31;
  const int wave = tid >> 5;
  const int nw   = blockDim.x >> 5;
  const int l15  = lane & 15;
  const int hi   = lane >> 4;
  const int Kw   = CinW * KS * KS;        // true weight row length

  union AB { v16bf v; unsigned short u[16]; };

  for (int mt = 0; mt < (Cout >> 4); ++mt) {
    const int m0 = mt << 4;

    // ---- hoist all A fragments (weights) for this m-tile into registers ---
    AB A[KSTEPS];
    #pragma unroll
    for (int kk = 0; kk < KSTEPS; ++kk) {
      const int g      = kk / CIN32;
      const int cibase = (kk - g * CIN32) * 32;
      const int kh     = (KS == 3) ? (g / 3) : 0;
      const int kw     = (KS == 3) ? (g - (g / 3) * 3) : 0;
      const int arow   = (m0 + l15) * Kw + kh * KS + kw;
      #pragma unroll
      for (int p = 0; p < 16; ++p) {
        int ci = cibase + p + (hi << 4);
        A[kk].u[p] = (ci < CinW) ? f2bf(W[arow + ci * (KS * KS)])
                                 : (unsigned short)0;
      }
    }

    // ---- pixel subtiles: wave handles 8 of the 32 -------------------------
    for (int s = wave; s < 32; s += nw) {
      const int hh = s >> 1;              // output row inside tile, 0..15
      const int cb = (s & 1) << 4;        // column base, 0 or 16
      v8f acc = {};

      #pragma unroll
      for (int kk = 0; kk < KSTEPS; ++kk) {
        const int g      = kk / CIN32;
        const int cibase = (kk - g * CIN32) * 32;
        const int kh     = (KS == 3) ? (g / 3) : 0;
        const int kw     = (KS == 3) ? (g - (g / 3) * 3) : 0;
        AB B;
        const unsigned short* bp = tile + (hh + kh) * WT + cb + l15 + kw;
        const int cB = cibase + (hi << 4);
        #pragma unroll
        for (int p = 0; p < 16; ++p)
          B.u[p] = bp[(cB + p) * HTWT];
        acc = __builtin_amdgcn_wmma_f32_16x16x32_bf16(false, A[kk].v, false, B.v,
                                                      (short)0, acc, false, false);
      }

      // ---- epilogue ----
      const int n = nbase + (r0 + hh) * 32 + cb + l15;
      #pragma unroll
      for (int i = 0; i < 8; ++i) {
        int m   = m0 + i + (hi << 3);
        float v = acc[i] + bias[m];
        int idx = m * NPIX + n;
        switch (mode) {
          case 0: out[idx] = f2bf(v); break;
          case 1: out[idx] = f2bf(v >= 0.f ? v : 0.01f * v); break;
          case 2: {
            float t = (v >= 0.f ? v : 0.01f * v) + bf2f(out[idx]);
            out[idx] = f2bf(t);
          } break;
          case 3: {
            float t = bng[m] * v * BN_S + bnb[m];
            t = (t >= 0.f ? t : 0.1f * t) * wgt[n >> 10];
            if (accum) t += bf2f(out[idx]);
            out[idx] = f2bf(t);
          } break;
          case 4: {
            float t = bng[m] * v * BN_S + bnb[m];
            out[idx] = f2bf(t > 0.f ? t : 0.f);
          } break;
          case 5: {
            float t = bng[m] * v * BN_S + bnb[m];
            outF32[(n >> 10) * 65536 + m * 1024 + (n & 1023)] = (t > 0.f ? t : 0.f);
          } break;
        }
      }
    }
  }
}

// ---------------------------------------------------------------------------
// Prepack for projection: dst[64][N] bf16 = concat(x[:,c0:c0+creg], y, zeros)
// ---------------------------------------------------------------------------
__global__ void prepack(const float* __restrict__ x, const float* __restrict__ y,
                        unsigned short* __restrict__ dst, int c0, int creg)
{
  int i = blockIdx.x * blockDim.x + threadIdx.x;
  if (i >= 64 * NPIX) return;
  int ch = i >> 17;            // / NPIX
  int n  = i & (NPIX - 1);
  int b  = n >> 10, hw = n & 1023;
  float v = 0.f;
  if (ch < creg)            v = x[b * 65536 + (c0 + ch) * 1024 + hw];
  else if (ch < creg + 32)  v = y[b * 32768 + (ch - creg) * 1024 + hw];
  dst[i] = f2bf(v);
}

// ---------------------------------------------------------------------------
// Per-expert input BatchNorm on bf16 activations.
// ---------------------------------------------------------------------------
__global__ void bn_apply(const unsigned short* __restrict__ src,
                         unsigned short* __restrict__ dst,
                         const float* __restrict__ g, const float* __restrict__ bta)
{
  int i = blockIdx.x * blockDim.x + threadIdx.x;
  if (i >= 64 * NPIX) return;
  int ch = i >> 17;
  dst[i] = f2bf(bf2f(src[i]) * BN_S * g[ch] + bta[ch]);
}

// ---------------------------------------------------------------------------
// Gates stage 1: GAP over H,W then the 3 gate MLPs -> logits[b][r][e]
// grid 128, block 64
// ---------------------------------------------------------------------------
struct GateP { const float* w1[3]; const float* b1[3];
               const float* w2[3]; const float* b2[3]; };

__global__ void gates1(const float* __restrict__ x, GateP gp,
                       float* __restrict__ logits)
{
  __shared__ float gap[64];
  __shared__ float h[64];
  const int b = blockIdx.x, t = threadIdx.x;
  float s = 0.f;
  const float* px = x + b * 65536 + t * 1024;
  for (int i = 0; i < 1024; ++i) s += px[i];
  gap[t] = s * (1.f / 1024.f);
  __syncthreads();
  for (int r = 0; r < 3; ++r) {
    float a = gp.b1[r][t];
    const float* w = gp.w1[r] + t * 64;
    for (int c = 0; c < 64; ++c) a += w[c] * gap[c];
    h[t] = a > 0.f ? a : 0.f;
    __syncthreads();
    if (t < 3) {
      float l = gp.b2[r][t];
      const float* w2 = gp.w2[r] + t * 64;
      for (int j = 0; j < 64; ++j) l += w2[j] * h[j];
      logits[b * 9 + r * 3 + t] = l;
    }
    __syncthreads();
  }
}

// ---------------------------------------------------------------------------
// Gates stage 2: softmax load sums -> aux loss; top-2 softmax -> wgt[r][e][b]
// one block, 128 threads (thread == batch index)
// ---------------------------------------------------------------------------
__global__ void gates2(const float* __restrict__ logits,
                       float* __restrict__ wgt, float* __restrict__ auxout)
{
  __shared__ float ls[9];
  const int b = threadIdx.x;
  if (b < 9) ls[b] = 0.f;
  __syncthreads();
  for (int r = 0; r < 3; ++r) {
    float l[3];
    l[0] = logits[b * 9 + r * 3 + 0];
    l[1] = logits[b * 9 + r * 3 + 1];
    l[2] = logits[b * 9 + r * 3 + 2];
    float m = fmaxf(l[0], fmaxf(l[1], l[2]));
    float e0 = __expf(l[0] - m), e1 = __expf(l[1] - m), e2 = __expf(l[2] - m);
    float sinv = 1.f / (e0 + e1 + e2);
    atomicAdd(&ls[r * 3 + 0], e0 * sinv);
    atomicAdd(&ls[r * 3 + 1], e1 * sinv);
    atomicAdd(&ls[r * 3 + 2], e2 * sinv);
    // top-2 with softmax over the two kept logits
    int i1 = 0; float v1 = l[0];
    if (l[1] > v1) { v1 = l[1]; i1 = 1; }
    if (l[2] > v1) { v1 = l[2]; i1 = 2; }
    int i2 = -1; float v2 = -3.4e38f;
    for (int e = 0; e < 3; ++e)
      if (e != i1 && l[e] > v2) { v2 = l[e]; i2 = e; }
    float c   = __expf(v2 - v1);
    float inv = 1.f / (1.f + c);
    float w[3] = {0.f, 0.f, 0.f};
    w[i1] = inv; w[i2] = c * inv;
    for (int e = 0; e < 3; ++e) wgt[(r * 3 + e) * 128 + b] = w[e];
  }
  __syncthreads();
  if (b == 0) {
    float aux = 0.f;
    for (int r = 0; r < 3; ++r) {
      float a0 = ls[r * 3], a1 = ls[r * 3 + 1], a2 = ls[r * 3 + 2];
      float mn = (a0 + a1 + a2) * (1.f / 3.f);
      float var = ((a0 - mn) * (a0 - mn) + (a1 - mn) * (a1 - mn) +
                   (a2 - mn) * (a2 - mn)) * 0.5f;          // ddof = 1
      aux += var / (mn * mn + 1e-10f);
    }
    auxout[0] = aux * 0.01f;
  }
}

// ---------------------------------------------------------------------------
// Host-side orchestration
// ---------------------------------------------------------------------------
extern "C" void kernel_launch(void* const* d_in, const int* in_sizes, int n_in,
                              void* d_out, int out_size, void* d_ws, size_t ws_size,
                              hipStream_t stream)
{
  (void)in_sizes; (void)n_in; (void)out_size; (void)ws_size;
  const float* x = (const float*)d_in[0];
  const float* y = (const float*)d_in[1];
  int i = 2;

  struct PP { const float *w, *b, *g, *beta; } proj[3];
  for (int r = 0; r < 3; ++r) {
    proj[r].w    = (const float*)d_in[i++];
    proj[r].b    = (const float*)d_in[i++];
    proj[r].g    = (const float*)d_in[i++];
    proj[r].beta = (const float*)d_in[i++];
  }
  struct SB { const float *w1, *b1, *w2, *b2, *wr, *br; };
  struct EX { const float *ing, *inb; SB sb[3];
              const float *fw, *fb, *fg, *fbeta; } ex[3][3];
  for (int r = 0; r < 3; ++r)
    for (int e = 0; e < 3; ++e) {
      EX& E = ex[r][e];
      E.ing = (const float*)d_in[i++];
      E.inb = (const float*)d_in[i++];
      for (int s = 0; s < 3; ++s) {
        E.sb[s].w1 = (const float*)d_in[i++];
        E.sb[s].b1 = (const float*)d_in[i++];
        E.sb[s].w2 = (const float*)d_in[i++];
        E.sb[s].b2 = (const float*)d_in[i++];
        E.sb[s].wr = (const float*)d_in[i++];
        E.sb[s].br = (const float*)d_in[i++];
      }
      E.fw = (const float*)d_in[i++];
      E.fb = (const float*)d_in[i++];
      E.fg = (const float*)d_in[i++];
      E.fbeta = (const float*)d_in[i++];
    }
  GateP gp;
  for (int r = 0; r < 3; ++r) {
    gp.w1[r] = (const float*)d_in[i++];
    gp.b1[r] = (const float*)d_in[i++];
    gp.w2[r] = (const float*)d_in[i++];
    gp.b2[r] = (const float*)d_in[i++];
  }
  struct FU { const float *w, *b, *g, *beta; } fu;
  fu.w = (const float*)d_in[i++]; fu.b = (const float*)d_in[i++];
  fu.g = (const float*)d_in[i++]; fu.beta = (const float*)d_in[i++];

  // ---- workspace carve (all bf16 raw u16 unless noted) ----
  char* ws = (char*)d_ws;
  unsigned short* REG  = (unsigned short*)ws;                          // [192][N] 48MB
  unsigned short* PROJ = (unsigned short*)(ws + ((size_t)48 << 20));   // 16MB
  unsigned short* X    = (unsigned short*)(ws + ((size_t)64 << 20));   // 16MB
  unsigned short* X2   = (unsigned short*)(ws + ((size_t)80 << 20));   // 16MB
  unsigned short* H1   = (unsigned short*)(ws + ((size_t)96 << 20));   // 8MB
  float* LOG = (float*)(ws + ((size_t)104 << 20));                     // [128][9]
  float* WGT = LOG + 128 * 9;                                          // [9][128]
  float* outF = (float*)d_out;

  gates1<<<128, 64, 0, stream>>>(x, gp, LOG);
  gates2<<<1, 128, 0, stream>>>(LOG, WGT, outF + 8388608);

  const int cs[3] = {21, 21, 22}, p0[3] = {0, 21, 42};
  const int EW_BLOCKS = (64 * NPIX + 255) / 256;
  const size_t LDS_1x1_64  = (size_t)64  * 16 * 32 * 2;   //  64 KB
  const size_t LDS_1x1_192 = (size_t)192 * 16 * 32 * 2;   // 192 KB
  const size_t LDS_3x3_64  = (size_t)64  * 18 * 34 * 2;   // ~76.5 KB
  const size_t LDS_3x3_32  = (size_t)32  * 18 * 34 * 2;   // ~38.3 KB

  for (int r = 0; r < 3; ++r) {
    // projection: prepack (pad Cin to 64) then 1x1 conv + BN + ReLU -> PROJ
    prepack<<<EW_BLOCKS, 256, 0, stream>>>(x, y, X2, p0[r], cs[r]);
    conv_wmma<1, 64><<<256, 128, LDS_1x1_64, stream>>>(
        X2, proj[r].w, proj[r].b, PROJ, nullptr,
        cs[r] + 32, 64, /*mode*/4, proj[r].g, proj[r].beta, nullptr, 0);

    for (int e = 0; e < 3; ++e) {
      EX& E = ex[r][e];
      bn_apply<<<EW_BLOCKS, 256, 0, stream>>>(PROJ, X, E.ing, E.inb);
      unsigned short* Xc = X;
      unsigned short* Xn = X2;
      for (int s = 0; s < 3; ++s) {
        // h1 = leaky(conv3x3(x))           64 -> 32
        conv_wmma<3, 64><<<256, 128, LDS_3x3_64, stream>>>(
            Xc, E.sb[s].w1, E.sb[s].b1, H1, nullptr,
            64, 32, /*mode*/1, nullptr, nullptr, nullptr, 0);
        // x2 = conv1x1(x) + br             64 -> 64 (residual path)
        conv_wmma<1, 64><<<256, 128, LDS_1x1_64, stream>>>(
            Xc, E.sb[s].wr, E.sb[s].br, Xn, nullptr,
            64, 64, /*mode*/0, nullptr, nullptr, nullptr, 0);
        // x2 += leaky(conv3x3(h1))         32 -> 64
        conv_wmma<3, 32><<<256, 128, LDS_3x3_32, stream>>>(
            H1, E.sb[s].w2, E.sb[s].b2, Xn, nullptr,
            32, 64, /*mode*/2, nullptr, nullptr, nullptr, 0);
        unsigned short* t = Xc; Xc = Xn; Xn = t;
      }
      // final 1x1 + BN + leaky(0.1), scaled by gate weight, accumulated
      conv_wmma<1, 64><<<256, 128, LDS_1x1_64, stream>>>(
          Xc, E.fw, E.fb, REG + (size_t)r * 64 * NPIX, nullptr,
          64, 64, /*mode*/3, E.fg, E.fbeta,
          WGT + (r * 3 + e) * 128, e > 0 ? 1 : 0);
    }
  }

  // fusion: 1x1 conv 192->64 over concatenated region outputs, BN+ReLU -> d_out
  conv_wmma<1, 192><<<256, 128, LDS_1x1_192, stream>>>(
      REG, fu.w, fu.b, nullptr, outF,
      192, 64, /*mode*/5, fu.g, fu.beta, nullptr, 0);
}
